// UV_Aggregator_33363305955573
// MI455X (gfx1250) — matrix operand order repaired and checked by reading, not compiled, not run on hardware
//
#include <hip/hip_runtime.h>

typedef _Float16 half_t;
typedef __attribute__((ext_vector_type(16))) _Float16 v16h;
typedef __attribute__((ext_vector_type(8)))  float    v8f;

#define B_NODES 4096
#define L_HIST  200
#define D       64
#define NT      13      // ceil(200/16) row tiles per node
#define WAVES   4
#define THREADS (WAVES * 32)
#define MAXT    4       // max tiles per wave = ceil(NT/WAVES)

// ---------------------------------------------------------------------------
// Fragment helpers (CDNA5 wave32 WMMA layouts, 16x16x32 f16)
// ---------------------------------------------------------------------------
union FragU { v16h v; float4 f[2]; };

__device__ __forceinline__ v8f wmma_f16(v16h a, v16h b, v8f c) {
  return __builtin_amdgcn_wmma_f32_16x16x32_f16(false, a, false, b, (short)0, c,
                                                false, false);
}

// Pack 16 f32 (four float4) into an f16 A-fragment register block
__device__ __forceinline__ v16h cvt16(float4 f0, float4 f1, float4 f2, float4 f3) {
  v16h r;
  r[0]  = (_Float16)f0.x; r[1]  = (_Float16)f0.y; r[2]  = (_Float16)f0.z; r[3]  = (_Float16)f0.w;
  r[4]  = (_Float16)f1.x; r[5]  = (_Float16)f1.y; r[6]  = (_Float16)f1.z; r[7]  = (_Float16)f1.w;
  r[8]  = (_Float16)f2.x; r[9]  = (_Float16)f2.y; r[10] = (_Float16)f2.z; r[11] = (_Float16)f2.w;
  r[12] = (_Float16)f3.x; r[13] = (_Float16)f3.y; r[14] = (_Float16)f3.z; r[15] = (_Float16)f3.w;
  return r;
}

// A-frag from LDS staging tile X[16][72] f16 (rows 144B, 16B aligned)
__device__ __forceinline__ v16h frag_a_lds(const half_t* Xp, int m, int koff, int k0) {
  FragU u;
  u.f[0] = *(const float4*)(Xp + m * 72 + k0 + koff);
  u.f[1] = *(const float4*)(Xp + m * 72 + k0 + 16 + koff);
  return u.v;
}

// B-frag from pre-transposed f16 weights W[N][Kstride] (WGP$-resident):
// lane j in 0..15 -> column (n*16+j), K = k0..k0+7, k0+16..k0+23 ; lanes 16..31 -> +8
__device__ __forceinline__ v16h frag_b16(const half_t* W, int kstride, int n,
                                         int k0, int lane) {
  const int j   = lane & 15;
  const int h16 = (lane >> 4) * 8;
  const half_t* p = W + (size_t)(n * 16 + j) * kstride + k0 + h16;
  FragU u;
  u.f[0] = *(const float4*)(p);
  u.f[1] = *(const float4*)(p + 16);
  return u.v;
}

// ---------------------------------------------------------------------------
// Kernel 1: transpose + f32->f16 the four weight matrices into ws
//   ws layout (half elements): W1t[64][128] @0, Wa1t[64][128] @8192,
//                              W2t[64][64] @16384, Wa2t[64][64] @20480
// ---------------------------------------------------------------------------
__global__ void prep_weights_kernel(const float* __restrict__ w_r1,
                                    const float* __restrict__ w_r2,
                                    const float* __restrict__ att1_w,
                                    const float* __restrict__ att2_w,
                                    half_t* __restrict__ Wt) {
  int i = blockIdx.x * blockDim.x + threadIdx.x;
  if (i < 8192) {
    int k = i >> 6, n = i & 63;
    Wt[n * 128 + k] = (half_t)w_r1[i];
  } else if (i < 16384) {
    int j = i - 8192; int k = j >> 6, n = j & 63;
    Wt[8192 + n * 128 + k] = (half_t)att1_w[j];
  } else if (i < 20480) {
    int j = i - 16384; int k = j >> 6, n = j & 63;
    Wt[16384 + n * 64 + k] = (half_t)w_r2[j];
  } else if (i < 24576) {
    int j = i - 20480; int k = j >> 6, n = j & 63;
    Wt[20480 + n * 64 + k] = (half_t)att2_w[j];
  }
}

// ---------------------------------------------------------------------------
// Kernel 2: fused aggregator. One workgroup per node (4 waves, 128 threads).
// Weight B-frags are deliberately re-loaded every tile (48 KB table lives in
// WGP$); a volatile asm makes the addresses loop-variant so the compiler
// can neither hoist ~380 VGPRs of weights nor spill them to scratch.
// ---------------------------------------------------------------------------
__global__ __launch_bounds__(THREADS, 4) void uv_aggregator_kernel(
    const int* __restrict__ nodes, const int* __restrict__ huv,
    const int* __restrict__ hr,
    const float* __restrict__ v2e, const float* __restrict__ u2e,
    const float* __restrict__ r2e,
    const float* __restrict__ b_r1, const float* __restrict__ b_r2,
    const float* __restrict__ att1_b, const float* __restrict__ att2_b,
    const float* __restrict__ att3_w, const float* __restrict__ att3_b,
    const half_t* __restrict__ Wt, float* __restrict__ out) {
  __shared__ __align__(16) half_t sO[208 * D];            // o_history, f16
  __shared__ __align__(16) half_t sStage[WAVES][16 * 72]; // per-wave staging
  __shared__ float sLogits[208];
  __shared__ float sRed[2 * WAVES];

  const int b    = blockIdx.x;
  const int tid  = threadIdx.x;
  const int wave = tid >> 5;
  const int lane = tid & 31;
  const int m    = lane & 15;
  const int hi   = lane >> 4;
  const int koff = hi * 8;
  const int c4   = hi * 2;   // float4-granular offset of this lane's chunks

  half_t* X = &sStage[wave][0];

  // ---- preload this wave's tile indices (breaks index->gather dep chain) ----
  int tiles_[MAXT], vids[MAXT], rids[MAXT];
  int myTiles = 0;
#pragma unroll
  for (int t = wave; t < NT; t += WAVES) {
    int rb = t * 16;
    int ll = rb + m;
    ll = (ll < L_HIST) ? ll : (L_HIST - 1);
    tiles_[myTiles] = t;
    vids[myTiles]   = huv[b * L_HIST + ll];
    rids[myTiles]   = hr[b * L_HIST + ll];
    ++myTiles;
  }
  {
    const float* p0 = v2e + (size_t)vids[0] * D;
    __builtin_prefetch(p0, 0, 3);
    __builtin_prefetch(p0 + 32, 0, 3);
  }

  // center-user embedding frags (attention input cols 64..127, broadcast rows)
  const float4* uvp = (const float4*)(u2e + (size_t)nodes[b] * D);
  const v16h auv2 = cvt16(uvp[c4], uvp[c4 + 1], uvp[c4 + 4], uvp[c4 + 5]);
  const v16h auv3 = cvt16(uvp[c4 + 8], uvp[c4 + 9], uvp[c4 + 12], uvp[c4 + 13]);

  for (int i = 0; i < myTiles; ++i) {
    const int tile    = tiles_[i];
    const int rowbase = tile * 16;
    const int l       = rowbase + m;

    // Opaque zero offset: defeats LICM of the weight-fragment loads so they
    // are re-fetched from WGP$ each tile instead of living in (spilled) VGPRs.
    int wofs = 0;
    asm volatile("" : "+v"(wofs));
    const half_t* W1t  = Wt + wofs;
    const half_t* Wa1t = Wt + 8192 + wofs;
    const half_t* W2t  = Wt + 16384 + wofs;
    const half_t* Wa2t = Wt + 20480 + wofs;

    // prefetch next tile's neighbor row while this tile computes
    if (i + 1 < myTiles) {
      const float* nev = v2e + (size_t)vids[i + 1] * D;
      __builtin_prefetch(nev, 0, 3);
      __builtin_prefetch(nev + 32, 0, 3);
    }

    // ---- gather: issue all 16 row loads as one batch, then convert ----
    const float4* evp = (const float4*)(v2e + (size_t)vids[i] * D);
    const float4* erp = (const float4*)(r2e + (size_t)rids[i] * D);
    float4 e0 = evp[c4],      e1 = evp[c4 + 1];
    float4 e2 = evp[c4 + 4],  e3 = evp[c4 + 5];
    float4 e4 = evp[c4 + 8],  e5 = evp[c4 + 9];
    float4 e6 = evp[c4 + 12], e7 = evp[c4 + 13];
    float4 g0 = erp[c4],      g1 = erp[c4 + 1];
    float4 g2 = erp[c4 + 4],  g3 = erp[c4 + 5];
    float4 g4 = erp[c4 + 8],  g5 = erp[c4 + 9];
    float4 g6 = erp[c4 + 12], g7 = erp[c4 + 13];
    const v16h a0 = cvt16(e0, e1, e2, e3);
    const v16h a1 = cvt16(e4, e5, e6, e7);
    const v16h a2 = cvt16(g0, g1, g2, g3);
    const v16h a3 = cvt16(g4, g5, g6, g7);

    // ---- message MLP layer 1: X1 = relu([e_uv;e_r] @ W1 + b1) ----
#pragma unroll
    for (int n = 0; n < 4; ++n) {
      v8f c = {};
      c = wmma_f16(a0, frag_b16(W1t, 128, n, 0, lane), c);
      c = wmma_f16(a1, frag_b16(W1t, 128, n, 32, lane), c);
      c = wmma_f16(a2, frag_b16(W1t, 128, n, 64, lane), c);
      c = wmma_f16(a3, frag_b16(W1t, 128, n, 96, lane), c);
      const float bias = b_r1[n * 16 + m];
      const int col = n * 16 + m;
#pragma unroll
      for (int r = 0; r < 8; ++r) {
        float v = c[r] + bias;
        X[(r + 8 * hi) * 72 + col] = (half_t)(v > 0.f ? v : 0.f);
      }
    }
    asm volatile("s_wait_dscnt 0" ::: "memory");

    // ---- layer 2: o = relu(X1 @ W2 + b2) ----
    {
      const v16h x0 = frag_a_lds(X, m, koff, 0);
      const v16h x1 = frag_a_lds(X, m, koff, 32);
      v8f oc[4];
#pragma unroll
      for (int n = 0; n < 4; ++n) {
        v8f c = {};
        c = wmma_f16(x0, frag_b16(W2t, 64, n, 0, lane), c);
        c = wmma_f16(x1, frag_b16(W2t, 64, n, 32, lane), c);
        const float bias = b_r2[n * 16 + m];
#pragma unroll
        for (int r = 0; r < 8; ++r) {
          float v = c[r] + bias;
          oc[n][r] = v > 0.f ? v : 0.f;
        }
      }
#pragma unroll
      for (int n = 0; n < 4; ++n) {
        const int col = n * 16 + m;
#pragma unroll
        for (int r = 0; r < 8; ++r) {
          const half_t h = (half_t)oc[n][r];
          const int row = r + 8 * hi;
          sO[(rowbase + row) * D + col] = h;  // keep for final aggregation
          X[row * 72 + col] = h;              // attention input cols 0..63
        }
      }
    }
    asm volatile("s_wait_dscnt 0" ::: "memory");

    // ---- attention layer 1: relu([o ; uv] @ Wa1 + ab1) ----
    {
      const v16h s0 = frag_a_lds(X, m, koff, 0);
      const v16h s1 = frag_a_lds(X, m, koff, 32);
#pragma unroll
      for (int n = 0; n < 4; ++n) {
        v8f c = {};
        c = wmma_f16(s0, frag_b16(Wa1t, 128, n, 0, lane), c);
        c = wmma_f16(s1, frag_b16(Wa1t, 128, n, 32, lane), c);
        c = wmma_f16(auv2, frag_b16(Wa1t, 128, n, 64, lane), c);
        c = wmma_f16(auv3, frag_b16(Wa1t, 128, n, 96, lane), c);
        const float bias = att1_b[n * 16 + m];
        const int col = n * 16 + m;
#pragma unroll
        for (int r = 0; r < 8; ++r) {
          float v = c[r] + bias;
          X[(r + 8 * hi) * 72 + col] = (half_t)(v > 0.f ? v : 0.f);
        }
      }
    }
    asm volatile("s_wait_dscnt 0" ::: "memory");

    // ---- attention layer 2: relu(a1 @ Wa2 + ab2) ----
    {
      const v16h t0 = frag_a_lds(X, m, koff, 0);
      const v16h t1 = frag_a_lds(X, m, koff, 32);
#pragma unroll
      for (int n = 0; n < 4; ++n) {
        v8f c = {};
        c = wmma_f16(t0, frag_b16(Wa2t, 64, n, 0, lane), c);
        c = wmma_f16(t1, frag_b16(Wa2t, 64, n, 32, lane), c);
        const float bias = att2_b[n * 16 + m];
        const int col = n * 16 + m;
#pragma unroll
        for (int r = 0; r < 8; ++r) {
          float v = c[r] + bias;
          X[(r + 8 * hi) * 72 + col] = (half_t)(v > 0.f ? v : 0.f);
        }
      }
    }
    asm volatile("s_wait_dscnt 0" ::: "memory");

    // ---- logits: a2 @ att3_w + b3  (one row per lane 0..15) ----
    if (lane < 16) {
      float acc = 0.f;
#pragma unroll 8
      for (int k = 0; k < D; ++k)
        acc += (float)X[m * 72 + k] * att3_w[k];
      const float lg = acc + att3_b[0];
      sLogits[l] = (l < L_HIST) ? lg : -1e30f;  // mask padded rows
    }
  }

  __syncthreads();

  // ---- softmax over L=200 neighbors ----
  float pm = -1e30f;
  for (int i = tid; i < L_HIST; i += THREADS) pm = fmaxf(pm, sLogits[i]);
#pragma unroll
  for (int off = 16; off; off >>= 1) pm = fmaxf(pm, __shfl_xor(pm, off, 32));
  if (lane == 0) sRed[wave] = pm;
  __syncthreads();
  const float mx = fmaxf(fmaxf(sRed[0], sRed[1]), fmaxf(sRed[2], sRed[3]));

  float ps = 0.f;
  for (int i = tid; i < L_HIST; i += THREADS) {
    const float e = __expf(sLogits[i] - mx);
    sLogits[i] = e;
    ps += e;
  }
#pragma unroll
  for (int off = 16; off; off >>= 1) ps += __shfl_xor(ps, off, 32);
  if (lane == 0) sRed[WAVES + wave] = ps;
  __syncthreads();
  const float inv = 1.f / (sRed[WAVES] + sRed[WAVES + 1] + sRed[WAVES + 2] + sRed[WAVES + 3]);

  // ---- to_feats[j] = sum_l att[l] * o[l][j] ----
  if (tid < D) {
    float acc = 0.f;
    for (int i = 0; i < L_HIST; ++i)
      acc += sLogits[i] * (float)sO[i * D + tid];
    out[(size_t)b * D + tid] = acc * inv;
  }
}

// ---------------------------------------------------------------------------
extern "C" void kernel_launch(void* const* d_in, const int* in_sizes, int n_in,
                              void* d_out, int out_size, void* d_ws, size_t ws_size,
                              hipStream_t stream) {
  (void)in_sizes; (void)n_in; (void)out_size; (void)ws_size;
  const int*   nodes  = (const int*)d_in[0];
  const int*   huv    = (const int*)d_in[1];
  const int*   hr     = (const int*)d_in[2];
  const float* v2e    = (const float*)d_in[3];
  const float* u2e    = (const float*)d_in[4];
  const float* r2e    = (const float*)d_in[5];
  const float* w_r1   = (const float*)d_in[6];
  const float* b_r1   = (const float*)d_in[7];
  const float* w_r2   = (const float*)d_in[8];
  const float* b_r2   = (const float*)d_in[9];
  const float* att1_w = (const float*)d_in[10];
  const float* att1_b = (const float*)d_in[11];
  const float* att2_w = (const float*)d_in[12];
  const float* att2_b = (const float*)d_in[13];
  const float* att3_w = (const float*)d_in[14];
  const float* att3_b = (const float*)d_in[15];
  half_t* Wt  = (half_t*)d_ws;
  float*  out = (float*)d_out;

  prep_weights_kernel<<<96, 256, 0, stream>>>(w_r1, w_r2, att1_w, att2_w, Wt);
  uv_aggregator_kernel<<<B_NODES, THREADS, 0, stream>>>(
      nodes, huv, hr, v2e, u2e, r2e, b_r1, b_r2, att1_b, att2_b, att3_w, att3_b,
      Wt, out);
}
